// MultiHeadAttention_56968446214675
// MI455X (gfx1250) — matrix operand order repaired and checked by reading, not compiled, not run on hardware
//
#include <hip/hip_runtime.h>

// ---------------------------------------------------------------------------
// MHA forward for MI455X (gfx1250), wave32.
// - v_wmma_f32_16x16x32_bf16 for all GEMMs (fp32 in -> bf16 RNE, f32 accum)
// - flash attention with double-buffered LDS K/V staging via
//   global_load_async_to_lds_b128 (ASYNCcnt) shared by all 4 waves of a block
// - global_prefetch in the projection GEMM k-loop
// Pipeline: gemm<0..2> (QKV proj -> bf16 ws) -> attn_flash -> gemm<3> (O proj)
// ---------------------------------------------------------------------------

typedef __bf16  v16bf __attribute__((ext_vector_type(16)));
typedef float   v8f   __attribute__((ext_vector_type(8)));

union FragBF {
    int4  i4[2];
    v16bf v;
};

__device__ __forceinline__ unsigned short f2bf(float f) {
    unsigned u = __float_as_uint(f);
    unsigned r = ((u >> 16) & 1u) + 0x7FFFu;   // round-to-nearest-even
    return (unsigned short)((u + r) >> 16);
}
__device__ __forceinline__ float bf2f(unsigned short h) {
    return __uint_as_float(((unsigned)h) << 16);
}
__device__ __forceinline__ v8f zero8() {
    v8f z = {0.f, 0.f, 0.f, 0.f, 0.f, 0.f, 0.f, 0.f};
    return z;
}

// --- CDNA5 async copy global->LDS (tracked by ASYNCcnt) --------------------
__device__ __forceinline__ void async_ld16(void* lds, const void* g) {
    asm volatile("global_load_async_to_lds_b128 %0, %1, off"
                 :: "v"((unsigned)(unsigned long long)lds),
                    "v"((unsigned long long)g)
                 : "memory");
}
__device__ __forceinline__ void wait_async_le4() {
    asm volatile("s_wait_asynccnt 0x4" ::: "memory");
}
__device__ __forceinline__ void wait_async_le0() {
    asm volatile("s_wait_asynccnt 0x0" ::: "memory");
}

static constexpr int Dm    = 1024;   // d_model
static constexpr int Mtot  = 4096;   // B*S
static constexpr int SEQ   = 2048;
static constexpr int NH    = 16;
static constexpr int BK    = 32;
static constexpr int PITCH = 40;     // GEMM LDS pitch in halves (80B)
static constexpr int KP    = 72;     // attn K-tile pitch in halves (144B)
static constexpr int VP    = 40;     // attn V-tile pitch in halves (80B)

// ---------------------------------------------------------------------------
// Tiled GEMM: D[m][n] = sum_k A[m][k] * W[n][k] + bias[n]
// MODE 0: A=f32 X, store bf16 Q  at [(b*16+h)*2048+s]*64+dk
// MODE 1: A=f32 X, store bf16 K  (same layout)
// MODE 2: A=f32 X, store bf16 V^T at [(b*16+h)*64+dk]*2048+s
// MODE 3: A=bf16 ws, store f32 to d_out[m*1024+n]
// ---------------------------------------------------------------------------
template <int MODE>
__global__ __launch_bounds__(128) void gemm_wmma(
    const void* __restrict__ Av, const float* __restrict__ W,
    const float* __restrict__ bias, void* __restrict__ dstv)
{
    __shared__ __align__(16) unsigned short sA[64 * PITCH];
    __shared__ __align__(16) unsigned short sB[64 * PITCH];

    const int m0   = blockIdx.x * 64;
    const int n0   = blockIdx.y * 64;
    const int tid  = threadIdx.x;
    const int w    = tid >> 5;
    const int lane = tid & 31;
    const int hi   = lane >> 4;
    const int lr   = lane & 15;

    v8f acc[4];
#pragma unroll
    for (int t = 0; t < 4; ++t) acc[t] = zero8();

    const float*          Af = (const float*)Av;
    const unsigned short* Ab = (const unsigned short*)Av;

    for (int k0 = 0; k0 < Dm; k0 += BK) {
        // cooperative load + f32->bf16 convert of A(64x32) and W(64x32) tiles
#pragma unroll
        for (int e = tid; e < 64 * BK; e += 128) {
            const int r = e >> 5, c = e & 31;
            float a;
            if (MODE == 3) a = bf2f(Ab[(size_t)(m0 + r) * Dm + k0 + c]);
            else           a = Af[(size_t)(m0 + r) * Dm + k0 + c];
            sA[r * PITCH + c] = f2bf(a);
            sB[r * PITCH + c] = f2bf(W[(size_t)(n0 + r) * Dm + k0 + c]);
        }
        // warm L2/L0 for the next k-tile (global_prefetch_b8)
        if (k0 + BK < Dm) {
            if (tid < 64) {
                if (MODE == 3) __builtin_prefetch(&Ab[(size_t)(m0 + tid) * Dm + k0 + BK], 0, 0);
                else           __builtin_prefetch(&Af[(size_t)(m0 + tid) * Dm + k0 + BK], 0, 0);
            } else {
                __builtin_prefetch(&W[(size_t)(n0 + tid - 64) * Dm + k0 + BK], 0, 0);
            }
        }
        __syncthreads();

        // A fragment (16x32 bf16): V0-3 <- k=8*hi.., V4-7 <- k=16+8*hi..
        FragBF a;
        const unsigned short* ar = &sA[(w * 16 + lr) * PITCH];
        a.i4[0] = *(const int4*)&ar[8 * hi];
        a.i4[1] = *(const int4*)&ar[16 + 8 * hi];

#pragma unroll
        for (int t = 0; t < 4; ++t) {
            FragBF bf;   // B frag (32x16): lane n=lr, k = 8*hi.. and 16+8*hi..
            const unsigned short* br = &sB[(t * 16 + lr) * PITCH];
            bf.i4[0] = *(const int4*)&br[8 * hi];
            bf.i4[1] = *(const int4*)&br[16 + 8 * hi];
            acc[t] = __builtin_amdgcn_wmma_f32_16x16x32_bf16(
                false, a.v, false, bf.v, (short)0, acc[t], false, false);
        }
        __syncthreads();
    }

    // C layout: lane -> n = lr, m = r + 8*hi
#pragma unroll
    for (int t = 0; t < 4; ++t) {
        const int   n  = n0 + t * 16 + lr;
        const float bv = bias[n];
#pragma unroll
        for (int r = 0; r < 8; ++r) {
            const int   m   = m0 + w * 16 + r + 8 * hi;
            const float val = acc[t][r] + bv;
            if (MODE == 3) {
                ((float*)dstv)[(size_t)m * Dm + n] = val;
            } else {
                unsigned short* d = (unsigned short*)dstv;
                const int bat = m >> 11, s = m & (SEQ - 1);
                const int hh  = n >> 6,  dk = n & 63;
                if (MODE == 2)
                    d[((size_t)(bat * NH + hh) * 64 + dk) * SEQ + s] = f2bf(val);
                else
                    d[((size_t)(bat * NH + hh) * SEQ + s) * 64 + dk] = f2bf(val);
            }
        }
    }
}

// ---------------------------------------------------------------------------
// Flash attention. Each wave owns 16 query rows; the block streams keys in
// chunks of 32 through double-buffered LDS tiles filled with async copies.
// Q,K: bf16 [b,h,s,dk]; Vt: bf16 [b,h,dk,s]; Out: bf16 [b,s,h*64+dk]
// ---------------------------------------------------------------------------
__device__ __forceinline__ void stage_kv(unsigned short* sKb, unsigned short* sVb,
                                         const unsigned short* Kh,
                                         const unsigned short* Vh,
                                         int kp, int tid)
{
    // K chunk: 32 rows x 64 halves -> sK (pitch 72); 2 x b128 per thread
    const int rk = tid >> 2, ck = (tid & 3) * 16;
    async_ld16(&sKb[rk * KP + ck],     &Kh[(size_t)(kp + rk) * 64 + ck]);
    async_ld16(&sKb[rk * KP + ck + 8], &Kh[(size_t)(kp + rk) * 64 + ck + 8]);
    // V^T chunk: 64 rows x 32 halves -> sV (pitch 40); 2 x b128 per thread
    const int rv = tid >> 1, cv = (tid & 1) * 16;
    async_ld16(&sVb[rv * VP + cv],     &Vh[(size_t)rv * SEQ + kp + cv]);
    async_ld16(&sVb[rv * VP + cv + 8], &Vh[(size_t)rv * SEQ + kp + cv + 8]);
}

__global__ __launch_bounds__(128) void attn_flash(
    const unsigned short* __restrict__ Q, const unsigned short* __restrict__ K,
    const unsigned short* __restrict__ Vt, const int* __restrict__ mask,
    unsigned short* __restrict__ Out)
{
    __shared__ __align__(16) unsigned short sK[2][32 * KP];
    __shared__ __align__(16) unsigned short sV[2][64 * VP];
    __shared__ __align__(16) float          sS[4][16 * 32];
    __shared__ __align__(16) unsigned short sP[4][16 * 32];
    __shared__ __align__(16) float          sRow[4][32];

    const int tid  = threadIdx.x;
    const int w    = tid >> 5;
    const int lane = tid & 31;
    const int hi   = lane >> 4;
    const int lr   = lane & 15;
    const int h    = blockIdx.y;
    const int b    = blockIdx.z;
    const int q0   = blockIdx.x * 64 + w * 16;

    const size_t headoff = (size_t)(b * NH + h) * SEQ * 64;
    const unsigned short* Qh = Q + headoff;
    const unsigned short* Kh = K + headoff;
    const unsigned short* Vh = Vt + headoff;
    const int* mb = mask + b * SEQ;

    // resident Q fragments for dk 0..31 / 32..63 (A layout)
    FragBF qf[2];
    {
        const unsigned short* qr = &Qh[(size_t)(q0 + lr) * 64];
        qf[0].i4[0] = *(const int4*)&qr[8 * hi];
        qf[0].i4[1] = *(const int4*)&qr[16 + 8 * hi];
        qf[1].i4[0] = *(const int4*)&qr[32 + 8 * hi];
        qf[1].i4[1] = *(const int4*)&qr[48 + 8 * hi];
    }

    v8f oacc[4];
#pragma unroll
    for (int t = 0; t < 4; ++t) oacc[t] = zero8();
    float m_i = -3.0e38f, l_i = 0.0f;

    constexpr int NJ = SEQ / 32;
    stage_kv(sK[0], sV[0], Kh, Vh, 0, tid);          // prologue: stage chunk 0

    for (int j = 0; j < NJ; ++j) {
        const int cur = j & 1;
        const int kp  = j * 32;
        // software pipeline: kick off chunk j+1 into the other buffer
        // (safe: end-of-iter barrier of j-1 means nobody still reads buf cur^1)
        if (j + 1 < NJ) {
            stage_kv(sK[cur ^ 1], sV[cur ^ 1], Kh, Vh, kp + 32, tid);
            wait_async_le4();        // async loads complete in order -> chunk j done
        } else {
            wait_async_le0();
        }
        __syncthreads();             // chunk j visible to all waves

        // mask loads issued early; latency hides under the QK WMMAs
        int4 mk[4];
        {
            const int* mrow = &mb[kp + 16 * hi];
#pragma unroll
            for (int g = 0; g < 4; ++g) mk[g] = *(const int4*)&mrow[4 * g];
        }

        // ---- scores S(16x32) = Q(16x64) . K^T ----
        // two independent accumulators + 4 pre-loaded B-frags -> no serial chain
        FragBF kf[2][2];
#pragma unroll
        for (int t = 0; t < 2; ++t)
#pragma unroll
            for (int c = 0; c < 2; ++c) {
                const unsigned short* kr = &sK[cur][(t * 16 + lr) * KP + c * 32];
                kf[t][c].i4[0] = *(const int4*)&kr[8 * hi];
                kf[t][c].i4[1] = *(const int4*)&kr[16 + 8 * hi];
            }
        v8f s0 = zero8(), s1 = zero8();
        s0 = __builtin_amdgcn_wmma_f32_16x16x32_bf16(
            false, qf[0].v, false, kf[0][0].v, (short)0, s0, false, false);
        s1 = __builtin_amdgcn_wmma_f32_16x16x32_bf16(
            false, qf[0].v, false, kf[1][0].v, (short)0, s1, false, false);
        s0 = __builtin_amdgcn_wmma_f32_16x16x32_bf16(
            false, qf[1].v, false, kf[0][1].v, (short)0, s0, false, false);
        s1 = __builtin_amdgcn_wmma_f32_16x16x32_bf16(
            false, qf[1].v, false, kf[1][1].v, (short)0, s1, false, false);
#pragma unroll
        for (int r = 0; r < 8; ++r) {
            sS[w][(r + 8 * hi) * 32 + lr]      = s0[r];
            sS[w][(r + 8 * hi) * 32 + 16 + lr] = s1[r];
        }
        __syncthreads();

        // ---- online softmax: lane handles (row=lr, cols 16*hi..16*hi+15) ----
        float sv[16];
        {
            const float* srow = &sS[w][lr * 32 + 16 * hi];
#pragma unroll
            for (int g = 0; g < 4; ++g) {
                const float4 fv = *(const float4*)&srow[4 * g];
                const int4   mv = mk[g];
                sv[4 * g + 0] = (mv.x == 0) ? -1.0e9f : fv.x * 0.125f;
                sv[4 * g + 1] = (mv.y == 0) ? -1.0e9f : fv.y * 0.125f;
                sv[4 * g + 2] = (mv.z == 0) ? -1.0e9f : fv.z * 0.125f;
                sv[4 * g + 3] = (mv.w == 0) ? -1.0e9f : fv.w * 0.125f;
            }
        }
        float cmax = sv[0];
#pragma unroll
        for (int i = 1; i < 16; ++i) cmax = fmaxf(cmax, sv[i]);
        cmax = fmaxf(cmax, __shfl_xor(cmax, 16, 32));
        const float mnew = fmaxf(m_i, cmax);

        float csum = 0.0f;
        __align__(16) unsigned short pv[16];
#pragma unroll
        for (int i = 0; i < 16; ++i) {
            const float p = __expf(sv[i] - mnew);
            csum += p;
            pv[i] = f2bf(p);
        }
        csum += __shfl_xor(csum, 16, 32);
        const float alpha = __expf(m_i - mnew);
        l_i = l_i * alpha + csum;
        m_i = mnew;

        {
            unsigned short* prow = &sP[w][lr * 32 + 16 * hi];
            *(int4*)&prow[0] = *(const int4*)&pv[0];
            *(int4*)&prow[8] = *(const int4*)&pv[8];
        }
        if (hi == 0) sRow[w][lr] = alpha;
        __syncthreads();

        // ---- rescale O accumulators by per-row alpha ----
        float al[8];
#pragma unroll
        for (int r = 0; r < 8; ++r) al[r] = sRow[w][r + 8 * hi];
#pragma unroll
        for (int t = 0; t < 4; ++t)
#pragma unroll
            for (int r = 0; r < 8; ++r) oacc[t][r] *= al[r];

        // ---- O(16x64) += P(16x32) . V(32x64) : batched B-frags, 4 WMMAs ----
        FragBF pf;
        {
            const unsigned short* pr = &sP[w][lr * 32];
            pf.i4[0] = *(const int4*)&pr[8 * hi];
            pf.i4[1] = *(const int4*)&pr[16 + 8 * hi];
        }
        FragBF vf[4];
#pragma unroll
        for (int t = 0; t < 4; ++t) {
            const unsigned short* vr = &sV[cur][(t * 16 + lr) * VP];
            vf[t].i4[0] = *(const int4*)&vr[8 * hi];
            vf[t].i4[1] = *(const int4*)&vr[16 + 8 * hi];
        }
#pragma unroll
        for (int t = 0; t < 4; ++t)
            oacc[t] = __builtin_amdgcn_wmma_f32_16x16x32_bf16(
                false, pf.v, false, vf[t].v, (short)0, oacc[t], false, false);
        __syncthreads();             // done reading sK/sV[cur], sS/sP this iter
    }

    // ---- finalize: O /= l, store bf16 in [b,s,h*64+dk] layout ----
    const float linv = 1.0f / fmaxf(l_i, 1.0e-20f);
    if (hi == 0) sRow[w][16 + lr] = linv;
    __syncthreads();
    float li[8];
#pragma unroll
    for (int r = 0; r < 8; ++r) li[r] = sRow[w][16 + r + 8 * hi];
#pragma unroll
    for (int t = 0; t < 4; ++t)
#pragma unroll
        for (int r = 0; r < 8; ++r) {
            const int m = q0 + r + 8 * hi;
            Out[(size_t)(b * SEQ + m) * Dm + h * 64 + t * 16 + lr] =
                f2bf(oacc[t][r] * li[r]);
        }
}

// ---------------------------------------------------------------------------
extern "C" void kernel_launch(void* const* d_in, const int* in_sizes, int n_in,
                              void* d_out, int out_size, void* d_ws, size_t ws_size,
                              hipStream_t stream)
{
    (void)in_sizes; (void)n_in; (void)out_size; (void)ws_size;
    const float* X    = (const float*)d_in[0];
    const int*   mask = (const int*)  d_in[1];
    const float* W_Q  = (const float*)d_in[2];
    const float* b_Q  = (const float*)d_in[3];
    const float* W_K  = (const float*)d_in[4];
    const float* b_K  = (const float*)d_in[5];
    const float* W_V  = (const float*)d_in[6];
    const float* b_V  = (const float*)d_in[7];
    const float* W_O  = (const float*)d_in[8];
    const float* b_O  = (const float*)d_in[9];

    unsigned short* q_ws  = (unsigned short*)d_ws;              // 8 MB each
    unsigned short* k_ws  = q_ws  + (size_t)Mtot * Dm;
    unsigned short* vt_ws = k_ws  + (size_t)Mtot * Dm;
    unsigned short* o_ws  = vt_ws + (size_t)Mtot * Dm;

    const dim3 gg(Mtot / 64, Dm / 64), bb(128);
    gemm_wmma<0><<<gg, bb, 0, stream>>>(X, W_Q, b_Q, q_ws);
    gemm_wmma<1><<<gg, bb, 0, stream>>>(X, W_K, b_K, k_ws);
    gemm_wmma<2><<<gg, bb, 0, stream>>>(X, W_V, b_V, vt_ws);
    attn_flash<<<dim3(SEQ / 64, NH, 2), dim3(128), 0, stream>>>(
        q_ws, k_ws, vt_ws, mask, o_ws);
    gemm_wmma<3><<<gg, bb, 0, stream>>>(o_ws, W_O, b_O, (float*)d_out);
}
